// CrossAttentionFusion_83657372991915
// MI455X (gfx1250) — matrix operand order repaired and checked by reading, not compiled, not run on hardware
//
#include <hip/hip_runtime.h>
#include <hip/hip_bf16.h>

typedef __attribute__((ext_vector_type(16))) _Float16 v16h;
typedef __attribute__((ext_vector_type(8)))  float    v8f;
typedef __attribute__((ext_vector_type(4)))  _Float16 v4h;

#define MDIM 8192
#define HDIM 2048

// GEMM tile configuration: 128x128 block tile, K-step 64, 8 waves (wave32).
#define BM 128
#define BN 128
#define BK 64
#define NKT (HDIM / BK)
#define LDK 72   // BK + 8 halves padding (144B rows, 16B aligned) to dodge bank conflicts

// C = A[M,K] * W[N,K]^T + bias[N], output f16, f32 accumulation via WMMA.
// A is f32 (A_F16=false) or f16 (A_F16=true); W is always f32 row-major [N,K].
// Double-buffered LDS: stage tile kt+1 while WMMAs consume tile kt; one barrier/iter.
template<bool A_F16>
__global__ __launch_bounds__(256)
void gemm_bias_f16out(const void* __restrict__ Ap,
                      const float* __restrict__ W,
                      const float* __restrict__ bias,
                      _Float16* __restrict__ Out)
{
    __shared__ _Float16 As[2][BM * LDK];
    __shared__ _Float16 Bs[2][BN * LDK];

    const int tid  = threadIdx.x;
    const int lane = tid & 31;
    const int wid  = tid >> 5;
    const int wm   = wid & 3;     // 4 waves along M (32 rows each)
    const int wn   = wid >> 2;    // 2 waves along N (64 cols each)
    const int rowBase = blockIdx.x * BM;
    const int colBase = blockIdx.y * BN;

    const int lhalf = lane >> 4;  // 0: lanes 0-15, 1: lanes 16-31
    const int l16   = lane & 15;

    const float*    Af = (const float*)Ap;
    const _Float16* Ah = (const _Float16*)Ap;

    // Stage one 128x64 A tile + 128x64 W tile (f32->f16) into LDS buffer `buf`.
    auto stage = [&](int buf, int k0) {
        if (A_F16) {
#pragma unroll
            for (int i = 0; i < 4; ++i) {
                int id = tid + i * 256;              // 1024 x 16-byte chunks
                int r  = id >> 3;
                int c  = (id & 7) * 8;
                *(uint4*)&As[buf][r * LDK + c] =
                    *(const uint4*)(Ah + (size_t)(rowBase + r) * HDIM + k0 + c);
            }
        } else {
#pragma unroll
            for (int i = 0; i < 8; ++i) {
                int id = tid + i * 256;              // 2048 x float4 chunks
                int r  = id >> 4;
                int c  = (id & 15) * 4;
                float4 f = *(const float4*)(Af + (size_t)(rowBase + r) * HDIM + k0 + c);
                v4h h;
                h.x = (_Float16)f.x; h.y = (_Float16)f.y;
                h.z = (_Float16)f.z; h.w = (_Float16)f.w;
                *(v4h*)&As[buf][r * LDK + c] = h;
            }
        }
#pragma unroll
        for (int i = 0; i < 8; ++i) {
            int id = tid + i * 256;
            int r  = id >> 4;
            int c  = (id & 15) * 4;
            float4 f = *(const float4*)(W + (size_t)(colBase + r) * HDIM + k0 + c);
            v4h h;
            h.x = (_Float16)f.x; h.y = (_Float16)f.y;
            h.z = (_Float16)f.z; h.w = (_Float16)f.w;
            *(v4h*)&Bs[buf][r * LDK + c] = h;
        }
        // gfx1250 global_prefetch_b8 of the tile after this one
        if (k0 + BK < HDIM) {
            int pr = tid >> 1;
            int pc = (tid & 1) * (BK / 2);
            if (A_F16)
                __builtin_prefetch(Ah + (size_t)(rowBase + pr) * HDIM + k0 + BK + pc, 0, 1);
            else
                __builtin_prefetch(Af + (size_t)(rowBase + pr) * HDIM + k0 + BK + pc, 0, 1);
            __builtin_prefetch(W + (size_t)(colBase + pr) * HDIM + k0 + BK + pc, 0, 1);
        }
    };

    v8f acc[2][4];
#pragma unroll
    for (int m = 0; m < 2; ++m)
#pragma unroll
        for (int n = 0; n < 4; ++n)
            acc[m][n] = (v8f)0.0f;

    stage(0, 0);
    __syncthreads();

    for (int kt = 0; kt < NKT; ++kt) {
        const int cur = kt & 1;
        // Stage next tile into the other buffer; no barrier between this and the
        // WMMA stream below, so global loads overlap the matrix pipe.
        if (kt + 1 < NKT)
            stage(cur ^ 1, (kt + 1) * BK);

#pragma unroll
        for (int kk = 0; kk < BK; kk += 32) {
            // A fragments: 16-bit 16x32 layout — lanes 0-15 hold M=l16, K={kk..kk+7, kk+16..kk+23};
            // lanes 16-31 hold K shifted by 8. Two ds_load_b128 each.
            union { v16h h; uint4 q[2]; } afr[2];
#pragma unroll
            for (int m = 0; m < 2; ++m) {
                int r = wm * 32 + m * 16 + l16;
                const _Float16* p = &As[cur][r * LDK + kk + lhalf * 8];
                afr[m].q[0] = *(const uint4*)p;
                afr[m].q[1] = *(const uint4*)(p + 16);
            }
            // B fragments: 32x16 (KxN); N=l16, same K order as A: lanes 0-15 K={0..7,16..23},
            // lanes 16-31 K shifted by 8.
            union { v16h h; unsigned u[8]; } bfr[4];
#pragma unroll
            for (int n = 0; n < 4; ++n) {
                int c = wn * 64 + n * 16 + l16;
                const _Float16* p = &Bs[cur][c * LDK + kk + lhalf * 8];
#pragma unroll
                for (int v = 0; v < 8; ++v)
                    bfr[n].u[v] = *(const unsigned*)(p + 2 * v + (v >> 2) * 8);
            }
#pragma unroll
            for (int m = 0; m < 2; ++m)
#pragma unroll
                for (int n = 0; n < 4; ++n)
                    acc[m][n] = __builtin_amdgcn_wmma_f32_16x16x32_f16(
                        false, afr[m].h, false, bfr[n].h,
                        (short)0, acc[m][n], false, false);
        }
        __syncthreads();
    }

    // Epilogue: C f32 layout — VGPR r: lanes 0-15 = (M=r, N=lane), lanes 16-31 = (M=8+r, N=lane-16)
#pragma unroll
    for (int n = 0; n < 4; ++n) {
        int col = colBase + wn * 64 + n * 16 + l16;
        float bv = bias[col];
#pragma unroll
        for (int m = 0; m < 2; ++m) {
            int rbase = rowBase + wm * 32 + m * 16 + lhalf * 8;
#pragma unroll
            for (int r = 0; r < 8; ++r) {
                float v = acc[m][n][r] + bv;
                Out[(size_t)(rbase + r) * HDIM + col] = (_Float16)v;
            }
        }
    }
}

// out[row, colOff + n] = LN(X[row] + E[row]) * g + b   (bias bo already folded into E)
__global__ __launch_bounds__(256)
void residual_ln(const float* __restrict__ X,
                 const _Float16* __restrict__ E,
                 const float* __restrict__ g,
                 const float* __restrict__ b,
                 float* __restrict__ out,
                 int colOff)
{
    __shared__ float red[256];
    __shared__ float red2[256];
    const int row = blockIdx.x;
    const int tid = threadIdx.x;

    float tv[8];
    float s = 0.f, sq = 0.f;
#pragma unroll
    for (int j = 0; j < 8; ++j) {
        int idx = j * 256 + tid;
        float t = X[(size_t)row * HDIM + idx] + (float)E[(size_t)row * HDIM + idx];
        tv[j] = t;
        s += t; sq += t * t;
    }
    red[tid] = s; red2[tid] = sq;
    __syncthreads();
    for (int off = 128; off > 0; off >>= 1) {
        if (tid < off) { red[tid] += red[tid + off]; red2[tid] += red2[tid + off]; }
        __syncthreads();
    }
    const float mean = red[0] * (1.0f / HDIM);
    const float var  = red2[0] * (1.0f / HDIM) - mean * mean;
    const float rstd = rsqrtf(var + 1e-5f);
#pragma unroll
    for (int j = 0; j < 8; ++j) {
        int idx = j * 256 + tid;
        out[(size_t)row * (2 * HDIM) + colOff + idx] =
            (tv[j] - mean) * rstd * g[idx] + b[idx];
    }
}

extern "C" void kernel_launch(void* const* d_in, const int* in_sizes, int n_in,
                              void* d_out, int out_size, void* d_ws, size_t ws_size,
                              hipStream_t stream) {
    const float* seq    = (const float*)d_in[0];
    const float* score  = (const float*)d_in[1];
    const float* in_w1  = (const float*)d_in[2];
    const float* in_b1  = (const float*)d_in[3];
    const float* out_w1 = (const float*)d_in[4];
    const float* out_b1 = (const float*)d_in[5];
    const float* in_w2  = (const float*)d_in[6];
    const float* in_b2  = (const float*)d_in[7];
    const float* out_w2 = (const float*)d_in[8];
    const float* out_b2 = (const float*)d_in[9];
    const float* ln1_g  = (const float*)d_in[10];
    const float* ln1_b  = (const float*)d_in[11];
    const float* ln2_g  = (const float*)d_in[12];
    const float* ln2_b  = (const float*)d_in[13];
    float* out = (float*)d_out;

    // softmax over a length-1 axis is identity: only the V slice of in_w matters.
    const float* wv1 = in_w1 + (size_t)2 * HDIM * HDIM;
    const float* bv1 = in_b1 + 2 * HDIM;
    const float* wv2 = in_w2 + (size_t)2 * HDIM * HDIM;
    const float* bv2 = in_b2 + 2 * HDIM;

    const size_t mh = (size_t)MDIM * HDIM;
    _Float16* V1 = (_Float16*)d_ws;   // 32 MB each
    _Float16* V2 = V1 + mh;
    _Float16* E1 = V2 + mh;
    _Float16* E2 = E1 + mh;

    dim3 gg(MDIM / BM, HDIM / BN), blk(256);
    // V_s = X_other @ Wv_s^T + bv_s
    gemm_bias_f16out<false><<<gg, blk, 0, stream>>>(score, wv1, bv1, V1);
    gemm_bias_f16out<false><<<gg, blk, 0, stream>>>(seq,   wv2, bv2, V2);
    // E_s = V_s @ Wo_s^T + bo_s
    gemm_bias_f16out<true ><<<gg, blk, 0, stream>>>(V1, out_w1, out_b1, E1);
    gemm_bias_f16out<true ><<<gg, blk, 0, stream>>>(V2, out_w2, out_b2, E2);
    // out = [ LN(seq + E1), LN(score + E2) ]
    residual_ln<<<dim3(MDIM), blk, 0, stream>>>(seq,   E1, ln1_g, ln1_b, out, 0);
    residual_ln<<<dim3(MDIM), blk, 0, stream>>>(score, E2, ln2_g, ln2_b, out, HDIM);
}
